// RPN_81509889343542
// MI455X (gfx1250) — matrix operand order repaired and checked by reading, not compiled, not run on hardware
//
#include <hip/hip_runtime.h>
#include <hip/hip_fp16.h>

// ---------------- problem constants ----------------
#define NB   2
#define H_   36
#define W_   120
#define HW_  4320            // H*W
#define CC   256
#define CK   2304            // 256*9
#define EPSV 1e-5f

typedef __attribute__((ext_vector_type(16))) _Float16 v16h;
typedef __attribute__((ext_vector_type(8)))  _Float16 v8h;
typedef __attribute__((ext_vector_type(8)))  float    v8f;

// B staging buffers use a K-chunked layout: element (n, k, p) lives at
//   ((n*(K/32) + k/32)*HW + p)*32 + k%32
// so a WMMA B fragment (fixed pixel, 16 consecutive k per half-wave) is one
// contiguous 32-byte-aligned v16h load.
__device__ __forceinline__ long chunkIdx(int n, int Kdim, int r, int p) {
    return ((((long)n * (Kdim >> 5) + (r >> 5)) * HW_ + p) << 5) + (r & 31);
}

// ---------------- helpers ----------------
__device__ __forceinline__ float bilin(const float* __restrict__ plane, float py, float px) {
    float y0f = floorf(py), x0f = floorf(px);
    int   y0  = (int)y0f,   x0  = (int)x0f;
    float wy1 = py - y0f,   wx1 = px - x0f;
    float wy0 = 1.f - wy1,  wx0 = 1.f - wx1;
    float v = 0.f;
    if ((unsigned)y0     < (unsigned)H_ && (unsigned)x0     < (unsigned)W_) v += plane[y0*W_ + x0]       * wy0 * wx0;
    if ((unsigned)y0     < (unsigned)H_ && (unsigned)(x0+1) < (unsigned)W_) v += plane[y0*W_ + x0+1]     * wy0 * wx1;
    if ((unsigned)(y0+1) < (unsigned)H_ && (unsigned)x0     < (unsigned)W_) v += plane[(y0+1)*W_ + x0]   * wy1 * wx0;
    if ((unsigned)(y0+1) < (unsigned)H_ && (unsigned)(x0+1) < (unsigned)W_) v += plane[(y0+1)*W_ + x0+1] * wy1 * wx1;
    return v;
}

// ---------------- weight pack: f32 (M,K) -> f16 (Mpad,K), zero padded rows ----------------
__global__ void packA_k(const float* __restrict__ A, int M, int Mpad, int K,
                        _Float16* __restrict__ Ap) {
    long i = (long)blockIdx.x * blockDim.x + threadIdx.x;
    if (i >= (long)Mpad * K) return;
    int row = (int)(i / K);
    Ap[i] = (row < M) ? (_Float16)A[i] : (_Float16)0.f;
}

// ---------------- f32 (NB,Kdim,HW) -> chunked f16 ----------------
__global__ void cvt_chunk_k(const float* __restrict__ src, int Kdim, _Float16* __restrict__ dst) {
    int p = blockIdx.x * blockDim.x + threadIdx.x;
    int c = blockIdx.y, n = blockIdx.z;
    if (p >= HW_) return;
    dst[chunkIdx(n, Kdim, c, p)] = (_Float16)src[((long)n * Kdim + c) * HW_ + p];
}

// ---------------- im2col (3x3 pad1) f32 -> chunked f16, rows (c*9+k) ----------------
__global__ void im2col3x3_k(const float* __restrict__ src, long srcStrideN, int C,
                            _Float16* __restrict__ dst) {
    int p = blockIdx.x * blockDim.x + threadIdx.x;
    int c = blockIdx.y;
    int n = blockIdx.z;
    if (p >= HW_) return;
    int y = p / W_, x = p - y * W_;
    const float* sp = src + (long)n * srcStrideN + (long)c * HW_;
#pragma unroll
    for (int k = 0; k < 9; ++k) {
        int sy = y + k / 3 - 1, sx = x + k % 3 - 1;
        float v = (sy >= 0 && sy < H_ && sx >= 0 && sx < W_) ? sp[sy * W_ + sx] : 0.f;
        dst[chunkIdx(n, C * 9, c * 9 + k, p)] = (_Float16)v;
    }
}

// ---------------- deformable im2col of depth: s rows (c*9+k), chunked f16 ----------------
__global__ void deform_s_k(const float* __restrict__ d,       // (N,256,HW)
                           const float* __restrict__ of,      // (N,27,HW), ch 0..17 offsets
                           _Float16* __restrict__ s) {
    int p = blockIdx.x * blockDim.x + threadIdx.x;
    int c = blockIdx.y, n = blockIdx.z;
    if (p >= HW_) return;
    int y = p / W_, x = p - y * W_;
    const float* plane = d + ((long)n * CC + c) * HW_;
    const float* ofp   = of + (long)n * 27 * HW_ + p;
#pragma unroll
    for (int k = 0; k < 9; ++k) {
        float dy = ofp[(long)(2 * k) * HW_];
        float dx = ofp[(long)(2 * k + 1) * HW_];
        float py = (float)(y + k / 3 - 1) + dy;
        float px = (float)(x + k % 3 - 1) + dx;
        s[chunkIdx(n, CK, c * 9 + k, p)] = (_Float16)bilin(plane, py, px);
    }
}

// ---------------- WMMA GEMM: C[n] = Ap(f16 Mpad x K) * B(chunked f16) ----------------
// Each wave: NT2 16-pixel tiles x MT 16-row tiles (MT*NT2 accumulators).
// Per k-step: NT2 B-fragment loads + MT A-fragment loads, then MT*NT2 wmmas
// (loads first so the scheduler can clause them and use partial waits).
// MODE 0: plain   1: +bias(e0)   2: BN(e0=g,e1=b,e2=m,e3=v)+ReLU   3: +Addt tensor
template <int MODE, int MT, int NT2>
__global__ void gemm16_wmma(const _Float16* __restrict__ Ap, int M, int K,
                            const _Float16* __restrict__ B,
                            float* __restrict__ C, long strideC,
                            const float* __restrict__ e0, const float* __restrict__ e1,
                            const float* __restrict__ e2, const float* __restrict__ e3,
                            const float* __restrict__ Addt, long strideAdd) {
    const int lane  = threadIdx.x;
    const int hi    = lane >> 4;
    const int lr    = lane & 15;
    const int col   = blockIdx.x * 16 * NT2 + lr;
    const int mbase = blockIdx.y * 16 * MT;
    const int n     = blockIdx.z;
    const int KC    = K >> 5;

    // B fragment base: fixed pixel column, 16 contiguous k (hi half) per chunk;
    // adjacent 16-pixel tile is +512 halfs within the same chunk.
    const _Float16* Bp = B + ((((long)n * KC) * HW_ + col) << 5) + hi * 16;
    // A fragment base: row = mbase + lr (+ 16*mt), two 8-half runs at +0 / +8
    const _Float16* Abase = Ap + (long)(mbase + lr) * K + hi * 16;

    v8f acc[MT][NT2];
    const v8f vzero = {0.f, 0.f, 0.f, 0.f, 0.f, 0.f, 0.f, 0.f};
#pragma unroll
    for (int mt = 0; mt < MT; ++mt)
#pragma unroll
        for (int t = 0; t < NT2; ++t) acc[mt][t] = vzero;

    for (int kc = 0; kc < KC; ++kc) {
        const _Float16* Bk = Bp + ((long)kc * HW_ << 5);
        v16h bfr[NT2];
#pragma unroll
        for (int t = 0; t < NT2; ++t)
            bfr[t] = *(const v16h*)(Bk + t * 512);
        if (kc + 1 < KC) __builtin_prefetch(Bp + (((long)kc + 1) * HW_ << 5), 0, 1);

        const _Float16* Ak = Abase + kc * 32;
        v16h afr[MT];
#pragma unroll
        for (int mt = 0; mt < MT; ++mt) {
            const _Float16* ar = Ak + (long)mt * 16 * K;
            v8h a0 = *(const v8h*)(ar);        // k = kb + 16*hi + [0,8)
            v8h a1 = *(const v8h*)(ar + 8);    // k = kb + 16*hi + 8 + [0,8)
            afr[mt] = __builtin_shufflevector(a0, a1, 0, 1, 2, 3, 4, 5, 6, 7,
                                              8, 9, 10, 11, 12, 13, 14, 15);
        }
#pragma unroll
        for (int mt = 0; mt < MT; ++mt)
#pragma unroll
            for (int t = 0; t < NT2; ++t)
                acc[mt][t] = __builtin_amdgcn_wmma_f32_16x16x32_f16(
                    false, afr[mt], false, bfr[t], (short)0, acc[mt][t], false, false);
    }

#pragma unroll
    for (int mt = 0; mt < MT; ++mt) {
#pragma unroll
        for (int v = 0; v < 8; ++v) {
            int row = mbase + mt * 16 + v + hi * 8;   // C/D layout: VGPR v -> M = v + 8*hi
            if (row < M) {
#pragma unroll
                for (int t = 0; t < NT2; ++t) {
                    int colt = col + t * 16;
                    float val = acc[mt][t][v];
                    if (MODE == 1) val += e0[row];
                    if (MODE == 2) {
                        val = (val - e2[row]) * rsqrtf(e3[row] + EPSV) * e0[row] + e1[row];
                        val = fmaxf(val, 0.f);
                    }
                    if (MODE == 3) val += Addt[(long)n * strideAdd + (long)row * HW_ + colt];
                    C[(long)n * strideC + (long)row * HW_ + colt] = val;
                }
            }
        }
    }
}

// ---------------- fused deform-sample(x) * sigmoid(filt) * softmax(aff) aggregation ----------------
__global__ void deform_agg_k(const float* __restrict__ cat,   // x at channels 0..255 of (N,1024,HW)
                             const float* __restrict__ offb,  // (N,18,HW)
                             const float* __restrict__ aff,   // (N,36,HW)
                             const float* __restrict__ of,    // (N,27,HW), filt at 18..26
                             float* __restrict__ out, int chOff) {
    int p = blockIdx.x * blockDim.x + threadIdx.x;
    int c = blockIdx.y, n = blockIdx.z;
    if (p >= HW_) return;
    int g = c >> 6;   // GROUP=4 -> 64 channels per group
    int y = p / W_, x = p - y * W_;

    const float* ap = aff + ((long)n * 36 + g * 9) * HW_ + p;
    float av[9];
    float mx = -3.0e38f;
#pragma unroll
    for (int k = 0; k < 9; ++k) { av[k] = ap[(long)k * HW_]; mx = fmaxf(mx, av[k]); }
    float ssum = 0.f;
#pragma unroll
    for (int k = 0; k < 9; ++k) { av[k] = __expf(av[k] - mx); ssum += av[k]; }
    float inv = 1.f / ssum;

    const float* fp    = of + ((long)n * 27 + 18) * HW_ + p;
    const float* op    = offb + (long)n * 18 * HW_ + p;
    const float* plane = cat + ((long)n * 1024 + c) * HW_;
    float o = 0.f;
#pragma unroll
    for (int k = 0; k < 9; ++k) {
        float fw = 1.f / (1.f + __expf(-fp[(long)k * HW_]));
        float dy = op[(long)(2 * k) * HW_];
        float dx = op[(long)(2 * k + 1) * HW_];
        float py = (float)(y + k / 3 - 1) + dy;
        float px = (float)(x + k % 3 - 1) + dx;
        o += bilin(plane, py, px) * fw * (av[k] * inv);
    }
    out[((long)n * 1024 + chOff + c) * HW_ + p] = o;
}

// ---------------- GroupNorm stats: one block per (n, group of 32 channels) ----------------
__global__ void gn_stats_k(const float* __restrict__ y, float* __restrict__ stats) {
    int gid = blockIdx.x;      // n*32 + g
    int n = gid >> 5, g = gid & 31;
    const float* base = y + ((long)n * 1024 + (long)g * 32) * HW_;
    float s = 0.f, s2 = 0.f;
    for (long i = threadIdx.x; i < 32L * HW_; i += blockDim.x) {
        float v = base[i];
        s += v; s2 += v * v;
    }
    __shared__ float sh[256], sh2[256];
    sh[threadIdx.x] = s; sh2[threadIdx.x] = s2;
    __syncthreads();
    for (int st = 128; st > 0; st >>= 1) {
        if ((int)threadIdx.x < st) { sh[threadIdx.x] += sh[threadIdx.x + st]; sh2[threadIdx.x] += sh2[threadIdx.x + st]; }
        __syncthreads();
    }
    if (threadIdx.x == 0) {
        float cnt  = 32.f * (float)HW_;
        float mean = sh[0] / cnt;
        float var  = sh2[0] / cnt - mean * mean;
        stats[gid * 2]     = mean;
        stats[gid * 2 + 1] = var;
    }
}

// ---------------- GN apply + ReLU + residual (in-place on y) ----------------
__global__ void gn_final_k(float* __restrict__ y, const float* __restrict__ stats,
                           const float* __restrict__ gg, const float* __restrict__ gb,
                           const float* __restrict__ rgb) {
    long i = (long)blockIdx.x * blockDim.x + threadIdx.x;
    long total = (long)NB * 1024 * HW_;
    if (i >= total) return;
    long c = (i / HW_) % 1024;
    long n = i / ((long)1024 * HW_);
    int  g = (int)(c >> 5);
    float m = stats[(n * 32 + g) * 2];
    float v = stats[(n * 32 + g) * 2 + 1];
    float val = (y[i] - m) * rsqrtf(v + EPSV) * gg[c] + gb[c];
    val = fmaxf(val, 0.f);
    y[i] = val + rgb[i];
}

// ---------------- launcher ----------------
extern "C" void kernel_launch(void* const* d_in, const int* in_sizes, int n_in,
                              void* d_out, int out_size, void* d_ws, size_t ws_size,
                              hipStream_t stream) {
    (void)in_sizes; (void)n_in; (void)out_size; (void)ws_size;
    // ---- inputs (setup_inputs() dict order, params walked in insertion order) ----
    const float* rgb      = (const float*)d_in[0];
    const float* depth[3] = {(const float*)d_in[1], (const float*)d_in[2], (const float*)d_in[3]};
    const float* smooth_w = (const float*)d_in[4];
    const float* sm_g = (const float*)d_in[5], *sm_b = (const float*)d_in[6];
    const float* sm_m = (const float*)d_in[7], *sm_v = (const float*)d_in[8];
    const float* w_off[3], *w_dconv[3], *w_conv1[3], *w_bg[3], *w_bb[3], *w_bm[3], *w_bv[3];
    for (int b = 0; b < 3; ++b) {
        int base = 9 + b * 7;
        w_off[b]   = (const float*)d_in[base + 0];
        w_dconv[b] = (const float*)d_in[base + 1];
        w_conv1[b] = (const float*)d_in[base + 2];
        w_bg[b]    = (const float*)d_in[base + 3];
        w_bb[b]    = (const float*)d_in[base + 4];
        w_bm[b]    = (const float*)d_in[base + 5];
        w_bv[b]    = (const float*)d_in[base + 6];
    }
    const float* offw[3]  = {(const float*)d_in[30], (const float*)d_in[31], (const float*)d_in[32]};
    const float* scale2_w = (const float*)d_in[33];
    const float* scale2_b = (const float*)d_in[34];
    const float* gn_g     = (const float*)d_in[35];
    const float* gn_b     = (const float*)d_in[36];
    float* out = (float*)d_out;

    // ---- workspace carve ----
    char* ws = (char*)d_ws;
    auto carve = [&](size_t bytes) { char* p = ws; ws += (bytes + 255) & ~(size_t)255; return p; };
    float*    cat   = (float*)   carve((size_t)NB * 1024 * HW_ * 4);   // [x | o1 | o2 | o3]
    _Float16* bufT  = (_Float16*)carve((size_t)NB * CK * HW_ * 2);     // chunked f16 staging
    float*    xw    = (float*)   carve((size_t)NB * CC * HW_ * 4);
    float*    ofb   = (float*)   carve((size_t)NB * 27 * HW_ * 4);
    float*    aff   = (float*)   carve((size_t)NB * 36 * HW_ * 4);
    float*    offs0 = (float*)   carve((size_t)NB * 18 * HW_ * 4);
    float*    offs1 = (float*)   carve((size_t)NB * 18 * HW_ * 4);
    float*    offs2 = (float*)   carve((size_t)NB * 18 * HW_ * 4);
    float*    stats = (float*)   carve((size_t)NB * 32 * 2 * 4);
    float*    offs[3] = {offs0, offs1, offs2};
    // packed f16 weights (zero-padded rows)
    _Float16* pSmooth = (_Float16*)carve((size_t)256 * 1024 * 2);
    _Float16* pScale2 = (_Float16*)carve((size_t)1024 * 1024 * 2);
    _Float16* pOffw[3], *pWoff[3], *pConv1[3], *pDconv[3];
    for (int b = 0; b < 3; ++b) {
        pOffw[b]  = (_Float16*)carve((size_t)32 * CK * 2);
        pWoff[b]  = (_Float16*)carve((size_t)32 * CK * 2);
        pConv1[b] = (_Float16*)carve((size_t)48 * CK * 2);
        pDconv[b] = (_Float16*)carve((size_t)256 * CK * 2);
    }

    const long sCat = 1024L * HW_, sC = (long)CC * HW_;
    const dim3 blkE(256), blkP(128);
    const dim3 gridPix((HW_ + 127) / 128, CC, NB);
    const int  NTB = HW_ / 32;   // 135 pixel-tile pairs (NT2 = 2)

    // ---- 0) pack all weights to padded f16 ----
    auto pack = [&](const float* A, int M, int Mpad, int K, _Float16* Ap) {
        long tot = (long)Mpad * K;
        packA_k<<<dim3((unsigned)((tot + 255) / 256)), blkE, 0, stream>>>(A, M, Mpad, K, Ap);
    };
    pack(smooth_w, 256, 256, 1024, pSmooth);
    pack(scale2_w, 1024, 1024, 1024, pScale2);
    for (int b = 0; b < 3; ++b) {
        pack(offw[b], 18, 32, CK, pOffw[b]);
        pack(w_off[b], 27, 32, CK, pWoff[b]);
        pack(w_conv1[b], 36, 48, CK, pConv1[b]);
        pack(w_dconv[b], 256, 256, CK, pDconv[b]);
    }

    // ---- 1) rgb -> chunked f16 ; smooth GEMM (M=256,K=1024) + BN + ReLU -> cat ch 0..255 ----
    cvt_chunk_k<<<dim3((HW_ + 127) / 128, 1024, NB), blkP, 0, stream>>>(rgb, 1024, bufT);
    gemm16_wmma<2, 4, 2><<<dim3(NTB, 4, NB), dim3(32), 0, stream>>>(
        pSmooth, 256, 1024, bufT, cat, sCat, sm_g, sm_b, sm_m, sm_v, nullptr, 0);

    // ---- 2) im2col(x) ; three offset convs on x (M=18, K=2304) ----
    im2col3x3_k<<<gridPix, blkP, 0, stream>>>(cat, sCat, CC, bufT);
    for (int b = 0; b < 3; ++b)
        gemm16_wmma<0, 2, 2><<<dim3(NTB, 1, NB), dim3(32), 0, stream>>>(
            pOffw[b], 18, CK, bufT, offs[b], 18L * HW_,
            nullptr, nullptr, nullptr, nullptr, nullptr, 0);

    // ---- 3) warp branches ----
    for (int b = 0; b < 3; ++b) {
        const float* d = depth[b];
        im2col3x3_k<<<gridPix, blkP, 0, stream>>>(d, sC, CC, bufT);
        gemm16_wmma<0, 2, 2><<<dim3(NTB, 1, NB), dim3(32), 0, stream>>>(
            pWoff[b], 27, CK, bufT, ofb, 27L * HW_,
            nullptr, nullptr, nullptr, nullptr, nullptr, 0);
        deform_s_k<<<gridPix, blkP, 0, stream>>>(d, ofb, bufT);
        gemm16_wmma<3, 4, 2><<<dim3(NTB, 4, NB), dim3(32), 0, stream>>>(
            pDconv[b], 256, CK, bufT, xw, sC,
            nullptr, nullptr, nullptr, nullptr, d, sC);
        im2col3x3_k<<<gridPix, blkP, 0, stream>>>(xw, sC, CC, bufT);
        gemm16_wmma<2, 3, 2><<<dim3(NTB, 1, NB), dim3(32), 0, stream>>>(
            pConv1[b], 36, CK, bufT, aff, 36L * HW_,
            w_bg[b], w_bb[b], w_bm[b], w_bv[b], nullptr, 0);
        deform_agg_k<<<gridPix, blkP, 0, stream>>>(cat, offs[b], aff, ofb, cat, 256 * (b + 1));
    }

    // ---- 4) scale2: y = scale2_w @ cat + bias (M=1024,K=1024) -> d_out ----
    cvt_chunk_k<<<dim3((HW_ + 127) / 128, 1024, NB), blkP, 0, stream>>>(cat, 1024, bufT);
    gemm16_wmma<1, 4, 2><<<dim3(NTB, 16, NB), dim3(32), 0, stream>>>(
        pScale2, 1024, 1024, bufT, out, sCat, scale2_b, nullptr, nullptr, nullptr, nullptr, 0);

    // ---- 5) GroupNorm(32) + ReLU + residual, in-place on d_out ----
    gn_stats_k<<<dim3(NB * 32), dim3(256), 0, stream>>>(out, stats);
    {
        long n = (long)NB * 1024 * HW_;
        gn_final_k<<<dim3((unsigned)((n + 255) / 256)), blkE, 0, stream>>>(out, stats, gn_g, gn_b, rgb);
    }
}
